// ResNetWSL_790273982738
// MI455X (gfx1250) — matrix operand order repaired and checked by reading, not compiled, not run on hardware
//
#include <hip/hip_runtime.h>
#include <hip/hip_bf16.h>
#include <math.h>

// ---------------------------------------------------------------------------
// Problem constants (from reference)
// ---------------------------------------------------------------------------
#define BATCH      64
#define FEAT_C     2048
#define HW2        196          // 14*14
#define NOUT       32           // NUM_CLASSES * NUM_MAPS
#define NCLS       8
#define NMAPS      4
#define KSTEPS     (FEAT_C / 32)   // 64 k-steps of 32 for wmma 16x16x32
#define NTILES     13              // ceil(196/16)

// workspace layout (bytes)
#define WP_OFF     0                                   // bf16 weights: 2*64*32*16 = 65536 elems
#define WP_BYTES   (2 * KSTEPS * 32 * 16 * 2)          // 131072
#define XC_OFF     WP_BYTES                            // xc: 64*32*196 f32
#define XC_BYTES   (BATCH * NOUT * HW2 * 4)            // 1605632
#define SAL_OFF    (XC_OFF + XC_BYTES)                 // sal: 64*196 f32

typedef __attribute__((ext_vector_type(16))) __bf16 v16bf;
typedef __attribute__((ext_vector_type(8)))  float  v8f;

__device__ __forceinline__ unsigned short f32_to_bf16_bits(float f) {
    union { float f; unsigned int u; } v; v.f = f;
    unsigned int u = v.u;
    u += 0x7FFFu + ((u >> 16) & 1u);     // round-to-nearest-even
    return (unsigned short)(u >> 16);
}

// ---------------------------------------------------------------------------
// Kernel 0: pre-swizzle down_w (fp32 [32,2048]) into a contiguous-k 16-bit
// A-matrix (16x32) per-lane layout, bf16, so the GEMM kernel does aligned
// 32-byte vector loads directly into the A operand.
// Layout: wp[((mt*64 + kk)*32 + lane)*16 + e]
//   m = lane&15, khalf = lane>>4, o = mt*16+m
//   e<8  -> K = khalf*16 + e         (VGPR0..3)
//   e>=8 -> K = khalf*16 + e         (VGPR4..7)
// ---------------------------------------------------------------------------
__global__ __launch_bounds__(256) void prep_w_kernel(const float* __restrict__ down_w,
                                                     unsigned short* __restrict__ wp) {
    int tid = blockIdx.x * blockDim.x + threadIdx.x;     // 65536 total
    int e    = tid & 15;
    int lane = (tid >> 4) & 31;
    int kk   = (tid >> 9) & 63;
    int mt   = tid >> 15;
    int m     = lane & 15;
    int khalf = lane >> 4;
    int o = mt * 16 + m;
    int c = kk * 32 + khalf * 16 + e;
    wp[tid] = f32_to_bf16_bits(down_w[o * FEAT_C + c]);
}

// ---------------------------------------------------------------------------
// Kernel 1: downconv GEMM via v_wmma_f32_16x16x32_bf16.
// grid = (64, 2), block = 256 (8 waves). Wave w handles n-tile
// nt = blockIdx.y*8 + w (13 tiles of 16 hw positions). Each wave computes
// both 16-row M tiles (o 0..15 and 16..31) sharing one B operand.
// B layout (32x16 bf16): lane = {khalf, n}; element e -> K = khalf*16 + e.
// D layout (16x16 f32): lane n = lane&15, vgpr r -> m = r + 8*(lane>>4).
//
// Boundary handling: lanes with hw >= 196 load from a clamped (valid)
// address with NO masking — WMMA output columns are independent, so the
// garbage lands only in D columns that are discarded at store time. This
// keeps every load unconditional (no per-lane EXEC juggling) and gives a
// single uniform loop body for all 13 n-tiles.
// ---------------------------------------------------------------------------
__global__ __launch_bounds__(256) void downconv_kernel(const float* __restrict__ x,
                                                       const unsigned short* __restrict__ wp,
                                                       float* __restrict__ xc) {
    const int b    = blockIdx.x;
    const int wid  = threadIdx.x >> 5;
    const int lane = threadIdx.x & 31;
    const int nt   = blockIdx.y * 8 + wid;
    if (nt >= NTILES) return;                    // wave-uniform: EXEC stays all-1s inside

    const int n     = lane & 15;
    const int khalf = lane >> 4;
    const int hw    = nt * 16 + n;
    const bool inb  = (hw < HW2);
    const int hws   = inb ? hw : 0;              // clamped address; value never masked

    const float* xb = x + (size_t)b * FEAT_C * HW2;
    const v16bf* a0p = (const v16bf*)(wp) + (size_t)(0 * KSTEPS) * 32;   // mt = 0
    const v16bf* a1p = (const v16bf*)(wp) + (size_t)(1 * KSTEPS) * 32;   // mt = 1

    v8f acc0 = {};
    v8f acc1 = {};

    for (int kk = 0; kk < KSTEPS; ++kk) {
        // Build B operand: column hw, K = kk*32 + khalf*16 + e  (fp32 -> bf16).
        // Pack two rounded high-halves per dword with one v_perm_b32.
        const float* xcol = xb + (size_t)(kk * 32 + khalf * 16) * HW2 + hws;
        union { v16bf v; unsigned int w[8]; } bu;
#pragma unroll
        for (int p = 0; p < 8; ++p) {
            float f0 = xcol[(size_t)(2 * p)     * HW2];
            float f1 = xcol[(size_t)(2 * p + 1) * HW2];
            unsigned int u0 = __builtin_bit_cast(unsigned int, f0) + 0x8000u; // round-half-up
            unsigned int u1 = __builtin_bit_cast(unsigned int, f1) + 0x8000u;
            bu.w[p] = __builtin_amdgcn_perm(u1, u0, 0x07060302u); // {hi16(u1),hi16(u0)}
        }
        v16bf bvec = bu.v;
        v16bf a0 = a0p[kk * 32 + lane];
        v16bf a1 = a1p[kk * 32 + lane];

        acc0 = __builtin_amdgcn_wmma_f32_16x16x32_bf16(
            false, a0, false, bvec, (short)0, acc0, false, false);
        acc1 = __builtin_amdgcn_wmma_f32_16x16x32_bf16(
            false, a1, false, bvec, (short)0, acc1, false, false);
    }

    if (inb) {
        float* xcb = xc + (size_t)b * NOUT * HW2;
#pragma unroll
        for (int r = 0; r < 8; ++r) {
            int m = r + 8 * khalf;
            xcb[(size_t)m * HW2 + hw]        = acc0[r];   // o = m
            xcb[(size_t)(16 + m) * HW2 + hw] = acc1[r];   // o = 16 + m
        }
    }
}

// ---------------------------------------------------------------------------
// Kernel 2: bias + GMP + class-wise pool + log_softmax (output 1) + saliency.
// sal[b,hw] = (1/32) * sum_o (xc[o,hw] + bias[o]) * x_out[o/4]
// ---------------------------------------------------------------------------
__global__ __launch_bounds__(256) void saliency_kernel(const float* __restrict__ xc,
                                                       const float* __restrict__ down_b,
                                                       float* __restrict__ out1,   // d_out[0..511]
                                                       float* __restrict__ sal) {
    const int b = blockIdx.x;
    const int t = threadIdx.x;
    __shared__ float bias_s[NOUT];
    __shared__ float gmp_s[NOUT];
    __shared__ float ls[NCLS];

    const float* xcb = xc + (size_t)b * NOUT * HW2;

    if (t < NOUT) bias_s[t] = down_b[t];
    __syncthreads();

    if (t < NOUT) {
        const float* p = xcb + (size_t)t * HW2;
        float m = -3.402823466e38f;
        for (int hw = 0; hw < HW2; ++hw) m = fmaxf(m, p[hw]);
        gmp_s[t] = m + bias_s[t];            // bias is constant per channel
    }
    __syncthreads();

    if (t == 0) {
        float sc[NCLS];
        float mx = -3.402823466e38f;
#pragma unroll
        for (int c = 0; c < NCLS; ++c) {
            float s = 0.25f * (gmp_s[c * 4] + gmp_s[c * 4 + 1] +
                               gmp_s[c * 4 + 2] + gmp_s[c * 4 + 3]);
            sc[c] = s;
            mx = fmaxf(mx, s);
        }
        float se = 0.0f;
#pragma unroll
        for (int c = 0; c < NCLS; ++c) se += __expf(sc[c] - mx);
        float lse = mx + __logf(se);
#pragma unroll
        for (int c = 0; c < NCLS; ++c) {
            float v = sc[c] - lse;
            ls[c] = v;
            out1[b * NCLS + c] = v;
        }
    }
    __syncthreads();

    for (int hw = t; hw < HW2; hw += blockDim.x) {
        float s = 0.0f;
#pragma unroll
        for (int o = 0; o < NOUT; ++o)
            s += (xcb[(size_t)o * HW2 + hw] + bias_s[o]) * ls[o >> 2];
        sal[b * HW2 + hw] = s * (1.0f / 32.0f);
    }
}

// ---------------------------------------------------------------------------
// Kernel 3: second pass over x (L2-resident). Folds concat + GAP + fc:
// logits[cls] = (1/196) * sum_{c,hw} (w1[cls,c] + w2[cls,c]*sal[hw]) * x[c,hw] + fc_b
// Wave owns channel rows (weights wave-uniform -> scalar loads); lanes stride hw.
// ---------------------------------------------------------------------------
__global__ __launch_bounds__(256) void gap_fc_kernel(const float* __restrict__ x,
                                                     const float* __restrict__ sal,
                                                     const float* __restrict__ fc_w,
                                                     const float* __restrict__ fc_b,
                                                     float* __restrict__ out2) { // d_out[512..1023]
    const int b    = blockIdx.x;
    const int wid  = threadIdx.x >> 5;
    const int lane = threadIdx.x & 31;

    __shared__ float sal_s[HW2];
    __shared__ float red[8][NCLS];

    for (int hw = threadIdx.x; hw < HW2; hw += blockDim.x)
        sal_s[hw] = sal[b * HW2 + hw];
    __syncthreads();

    const float* xb = x + (size_t)b * FEAT_C * HW2;
    float acc[NCLS];
#pragma unroll
    for (int c = 0; c < NCLS; ++c) acc[c] = 0.0f;

    for (int ch = wid; ch < FEAT_C; ch += 8) {
        const float* row = xb + (size_t)ch * HW2;
        // prefetch next row for this wave (L2 hit expected; hides latency)
        if (ch + 8 < FEAT_C)
            __builtin_prefetch(row + (size_t)8 * HW2 + lane, 0, 1);

        float w1c[NCLS], w2c[NCLS];
#pragma unroll
        for (int c = 0; c < NCLS; ++c) {
            w1c[c] = fc_w[c * (2 * FEAT_C) + ch];            // wave-uniform
            w2c[c] = fc_w[c * (2 * FEAT_C) + FEAT_C + ch];
        }
        for (int hw = lane; hw < HW2; hw += 32) {
            float xv = row[hw];
            float s  = sal_s[hw];
#pragma unroll
            for (int c = 0; c < NCLS; ++c)
                acc[c] = fmaf(fmaf(w2c[c], s, w1c[c]), xv, acc[c]);
        }
    }

    // wave32 tree reduction
#pragma unroll
    for (int off = 16; off > 0; off >>= 1) {
#pragma unroll
        for (int c = 0; c < NCLS; ++c)
            acc[c] += __shfl_down(acc[c], off, 32);
    }
    if (lane == 0) {
#pragma unroll
        for (int c = 0; c < NCLS; ++c) red[wid][c] = acc[c];
    }
    __syncthreads();

    if (threadIdx.x == 0) {
        float lg[NCLS];
        float mx = -3.402823466e38f;
#pragma unroll
        for (int c = 0; c < NCLS; ++c) {
            float s = 0.0f;
#pragma unroll
            for (int w = 0; w < 8; ++w) s += red[w][c];
            s = s * (1.0f / (float)HW2) + fc_b[c];
            lg[c] = s;
            mx = fmaxf(mx, s);
        }
        float se = 0.0f;
#pragma unroll
        for (int c = 0; c < NCLS; ++c) se += __expf(lg[c] - mx);
        float lse = mx + __logf(se);
#pragma unroll
        for (int c = 0; c < NCLS; ++c)
            out2[b * NCLS + c] = lg[c] - lse;
    }
}

// ---------------------------------------------------------------------------
// Launch
// ---------------------------------------------------------------------------
extern "C" void kernel_launch(void* const* d_in, const int* in_sizes, int n_in,
                              void* d_out, int out_size, void* d_ws, size_t ws_size,
                              hipStream_t stream) {
    const float* x      = (const float*)d_in[0];   // [64,2048,14,14]
    const float* down_w = (const float*)d_in[1];   // [32,2048]
    const float* down_b = (const float*)d_in[2];   // [32]
    const float* fc_w   = (const float*)d_in[3];   // [8,4096]
    const float* fc_b   = (const float*)d_in[4];   // [8]
    float* out = (float*)d_out;                    // [512 x_out | 512 x_conv_out]

    unsigned short* wp  = (unsigned short*)((char*)d_ws + WP_OFF);
    float*          xc  = (float*)((char*)d_ws + XC_OFF);
    float*          sal = (float*)((char*)d_ws + SAL_OFF);

    (void)in_sizes; (void)n_in; (void)out_size; (void)ws_size;

    // 0) weight swizzle+cvt: 65536 elems
    prep_w_kernel<<<256, 256, 0, stream>>>(down_w, wp);

    // 1) downconv GEMM (WMMA bf16), pass 1 over x
    downconv_kernel<<<dim3(BATCH, 2), 256, 0, stream>>>(x, wp, xc);

    // 2) GMP + log_softmax (output 1) + saliency map
    saliency_kernel<<<BATCH, 256, 0, stream>>>(xc, down_b, out, sal);

    // 3) fused concat/GAP/fc + log_softmax (output 2), pass 2 over x
    gap_fc_kernel<<<BATCH, 256, 0, stream>>>(x, sal, fc_w, fc_b, out + BATCH * NCLS);
}
